// WindowAttention_35966056137265
// MI455X (gfx1250) — matrix operand order repaired and checked by reading, not compiled, not run on hardware
//
#include <hip/hip_runtime.h>
#include <hip/hip_bf16.h>

typedef __attribute__((ext_vector_type(16))) _Float16 v16h;
typedef __attribute__((ext_vector_type(8)))  _Float16 v8h;
typedef __attribute__((ext_vector_type(8)))  float    v8f;

#define TOK   49
#define PADT  64
#define EMB   128
#define HEADS 4
#define HD    32
#define NWIN  64
#define NBLK  4096

// ---- WMMA operand helpers -------------------------------------------------
// A operand (16x32 f16): lane holds K = {kb..kb+7, kb+16..kb+23}; caller folds
// the +8 upper-half-wave offset into the pointer. Two 16B chunks at p, p+16.
__device__ __forceinline__ v16h ldA(const _Float16* p) {
  v8h lo = *(const v8h*)p;
  v8h hi = *(const v8h*)(p + 16);
  v16h r;
#pragma unroll
  for (int i = 0; i < 8; ++i) { r[i] = lo[i]; r[i + 8] = hi[i]; }
  return r;
}

// B operand (32x16 f16) from [N][K] storage: lane holds K = {kb..kb+7, kb+16..kb+23}
// at its column (caller folds the +8 upper-half-wave offset into the pointer).
__device__ __forceinline__ v16h ldB(const _Float16* p) {
  v8h lo = *(const v8h*)p;
  v8h hi = *(const v8h*)(p + 16);
  v16h r;
#pragma unroll
  for (int i = 0; i < 8; ++i) { r[i] = lo[i]; r[i + 8] = hi[i]; }
  return r;
}

__device__ __forceinline__ v8f wmma_f16(v16h a, v16h b, v8f c) {
  return __builtin_amdgcn_wmma_f32_16x16x32_f16(
      /*neg_a=*/false, a, /*neg_b=*/false, b,
      /*c_mod=*/(short)0, c, /*reuse_a=*/false, /*reuse_b=*/false);
}

// ---- Kernel 0: transpose + f16-convert weights into workspace -------------
// qkv_w  (128x384 f32, [k][n]) -> qkv_wt  (f16 [n][k], 384x128)
// proj_w (128x128 f32, [k][n]) -> proj_wt (f16 [n][k], 128x128)
__global__ void swin_prep_weights(const float* __restrict__ qkv_w,
                                  const float* __restrict__ proj_w,
                                  _Float16* __restrict__ qkv_wt,
                                  _Float16* __restrict__ proj_wt) {
  int idx = blockIdx.x * 256 + threadIdx.x;
  const int NQ = EMB * 3 * EMB;      // 49152
  const int NP = EMB * EMB;          // 16384
  if (idx < NQ) {
    int k = idx / (3 * EMB), n = idx % (3 * EMB);
    qkv_wt[n * EMB + k] = (_Float16)qkv_w[idx];
  } else if (idx < NQ + NP) {
    int j = idx - NQ;
    int k = j / EMB, n = j % EMB;
    proj_wt[n * EMB + k] = (_Float16)proj_w[j];
  }
}

// ---- Main fused kernel: one block (8 waves) per window --------------------
__global__ __launch_bounds__(256)
void swin_window_attention(const float* __restrict__ x,
                           const _Float16* __restrict__ qkv_wt,
                           const float* __restrict__ qkv_b,
                           const _Float16* __restrict__ proj_wt,
                           const float* __restrict__ proj_b,
                           const float* __restrict__ bias_table,
                           const float* __restrict__ mask,
                           const int* __restrict__ rel_index,
                           float* __restrict__ out) {
  __shared__ _Float16 lds_x [PADT * EMB];        // 16 KB: x (f16), later O
  __shared__ _Float16 lds_q [HEADS * PADT * HD]; // 16 KB: Q pre-scaled, [h][tok][d]
  __shared__ _Float16 lds_k [HEADS * PADT * HD]; // 16 KB: K, [h][tok][d]  (B layout for QK^T)
  __shared__ _Float16 lds_vt[HEADS * HD * PADT]; // 16 KB: V^T, [h][d][tok] (B layout for PV)
  __shared__ _Float16 lds_sp[PADT * PADT];       //  8 KB: S then P for one head

  const int tid  = threadIdx.x;
  // Force wave id into an SGPR so every tile index derived from it is scalar:
  // tile selection then compiles to s_cbranch / SALU, not exec-mask divergence.
  const int wave = __builtin_amdgcn_readfirstlane(tid >> 5);
  const int lane = tid & 31;
  const int l15  = lane & 15;
  const int lh   = lane >> 4;   // half-wave: 0 or 1
  const int aOff = lh * 8;      // A-operand K offset for upper half
  const int bOff = lh * 8;      // B-operand K offset for upper half
  const int cOff = lh * 8;      // C/D row offset for upper half
  const int b    = blockIdx.x;
  const int widx = b & (NWIN - 1);

  // ---- stage x as f16, zero-pad rows [49,64) ----
  const float* xb = x + (size_t)b * TOK * EMB;
  for (int i = tid; i < PADT * EMB; i += 256) {
    int row = i >> 7, col = i & 127;
    lds_x[i] = (_Float16)(row < TOK ? xb[row * EMB + col] : 0.0f);
  }
  __syncthreads();

  // ---- Phase 1: qkv = x @ qkv_w + qkv_b ; scatter into Q/K/V^T LDS ----
  for (int t = wave; t < 4 * 24; t += 8) {
    int mt = t / 24, nt = t % 24;          // scalar (t is SGPR-derived)
    int n  = nt * 16 + l15;                // output column in [0,384)
    v8f acc = {};
#pragma unroll
    for (int kk = 0; kk < 4; ++kk) {
      v16h a  = ldA(lds_x + (mt * 16 + l15) * EMB + kk * 32 + aOff);
      v16h bm = ldB(qkv_wt + n * EMB + kk * 32 + bOff);
      acc = wmma_f16(a, bm, acc);
    }
    float qb = qkv_b[n];
    // A 16-wide tile never crosses a 128-col boundary -> part/head are
    // functions of nt only (scalar, wave-uniform). Branch hoisted out of the
    // per-row store loop: one scalar branch per tile.
    int part = nt >> 3;                    // 0=q 1=k 2=v
    int hd   = (nt & 7) >> 1;              // head index
    int dm   = (nt & 1) * 16 + l15;        // dim within head
    int rb   = mt * 16 + cOff;             // first row this lane writes
    if (part == 0) {
      _Float16* dst = lds_q + (hd * PADT + rb) * HD + dm;
#pragma unroll
      for (int r = 0; r < 8; ++r)
        dst[r * HD] = (_Float16)((acc[r] + qb) * 0.17677669529663688f);
    } else if (part == 1) {
      _Float16* dst = lds_k + (hd * PADT + rb) * HD + dm;
#pragma unroll
      for (int r = 0; r < 8; ++r)
        dst[r * HD] = (_Float16)(acc[r] + qb);
    } else {
      // V^T layout: the 8 rows a lane writes are contiguous -> one b128 store.
      v8h pk;
#pragma unroll
      for (int r = 0; r < 8; ++r)
        pk[r] = (_Float16)(acc[r] + qb);
      *(v8h*)(lds_vt + (hd * HD + dm) * PADT + rb) = pk;
    }
  }
  __syncthreads();

  // ---- Phase 2: per-head attention ----
  for (int h = 0; h < HEADS; ++h) {
    // S = Q_h @ K_h^T : 16 tiles, 2 per wave, one WMMA each (K = 32 = head dim)
#pragma unroll
    for (int tt = 0; tt < 2; ++tt) {
      int t  = wave * 2 + tt;
      int mt = t >> 2, nt = t & 3;         // scalar
      v16h a  = ldA(lds_q + (h * PADT + mt * 16 + l15) * HD + aOff);
      v16h bm = ldB(lds_k + (h * PADT + nt * 16 + l15) * HD + bOff);
      v8f acc = {};
      acc = wmma_f16(a, bm, acc);
      int j = nt * 16 + l15;
#pragma unroll
      for (int r = 0; r < 8; ++r)
        lds_sp[(mt * 16 + r + cOff) * PADT + j] = (_Float16)acc[r];
    }
    __syncthreads();

    // softmax: one thread per valid row, bias + shift-mask added on the fly
    if (tid < TOK) {
      int i = tid;
      const int*   ri   = rel_index + i * TOK;
      const float* mrow = mask + ((size_t)widx * TOK + i) * TOK;
      float mx = -1e30f;
      for (int j = 0; j < TOK; ++j) {
        float s = (float)lds_sp[i * PADT + j] + bias_table[ri[j] * HEADS + h] + mrow[j];
        lds_sp[i * PADT + j] = (_Float16)s;
        mx = fmaxf(mx, s);
      }
      float sum = 0.0f;
      for (int j = 0; j < TOK; ++j) {
        float e = __expf((float)lds_sp[i * PADT + j] - mx);
        sum += e;
        lds_sp[i * PADT + j] = (_Float16)e;
      }
      float inv = 1.0f / sum;
      for (int j = 0; j < TOK; ++j)
        lds_sp[i * PADT + j] = (_Float16)((float)lds_sp[i * PADT + j] * inv);
      for (int j = TOK; j < PADT; ++j)
        lds_sp[i * PADT + j] = (_Float16)0.0f;   // kill padded K-columns for PV
    }
    __syncthreads();

    // O_h = P @ V_h : 8 tiles, 1 per wave, K = 64 tokens -> 2 WMMA k-steps
    {
      int mt = wave >> 1, nt2 = wave & 1;  // scalar
      v8f acc = {};
#pragma unroll
      for (int kk = 0; kk < 2; ++kk) {
        v16h a  = ldA(lds_sp + (mt * 16 + l15) * PADT + kk * 32 + aOff);
        v16h bm = ldB(lds_vt + (h * HD + nt2 * 16 + l15) * PADT + kk * 32 + bOff);
        acc = wmma_f16(a, bm, acc);
      }
      int cc = h * HD + nt2 * 16 + l15;
#pragma unroll
      for (int r = 0; r < 8; ++r)
        lds_x[(mt * 16 + r + cOff) * EMB + cc] = (_Float16)acc[r];  // x buffer reused as O
    }
    __syncthreads();
  }

  // ---- Phase 3: out = O @ proj_w + proj_b ----
  for (int t = wave; t < 4 * 8; t += 8) {
    int mt = t >> 3, nt = t & 7;           // scalar
    int n  = nt * 16 + l15;
    v8f acc = {};
#pragma unroll
    for (int kk = 0; kk < 4; ++kk) {
      v16h a  = ldA(lds_x + (mt * 16 + l15) * EMB + kk * 32 + aOff);
      v16h bm = ldB(proj_wt + n * EMB + kk * 32 + bOff);
      acc = wmma_f16(a, bm, acc);
    }
    float pb = proj_b[n];
#pragma unroll
    for (int r = 0; r < 8; ++r) {
      int row = mt * 16 + r + cOff;
      if (row < TOK)
        out[((size_t)b * TOK + row) * EMB + n] = acc[r] + pb;
    }
  }
}

extern "C" void kernel_launch(void* const* d_in, const int* in_sizes, int n_in,
                              void* d_out, int out_size, void* d_ws, size_t ws_size,
                              hipStream_t stream) {
  const float* x          = (const float*)d_in[0];
  const float* qkv_w      = (const float*)d_in[1];
  const float* qkv_b      = (const float*)d_in[2];
  const float* proj_w     = (const float*)d_in[3];
  const float* proj_b     = (const float*)d_in[4];
  const float* bias_table = (const float*)d_in[5];
  const float* mask       = (const float*)d_in[6];
  const int*   rel_index  = (const int*)d_in[7];
  float* out = (float*)d_out;

  // workspace layout: f16 transposed weights
  _Float16* qkv_wt  = (_Float16*)d_ws;                       // 384*128*2 = 98304 B
  _Float16* proj_wt = (_Float16*)((char*)d_ws + 98304);      // 128*128*2 = 32768 B

  const int prep_elems = EMB * 3 * EMB + EMB * EMB;          // 65536
  swin_prep_weights<<<(prep_elems + 255) / 256, 256, 0, stream>>>(
      qkv_w, proj_w, qkv_wt, proj_wt);

  swin_window_attention<<<NBLK, 256, 0, stream>>>(
      x, qkv_wt, qkv_b, proj_wt, proj_b, bias_table, mask, rel_index, out);
}
